// S5SSM_22591527977585
// MI455X (gfx1250) — compile-verified, matrix operand order checked by reading, not run on hardware
//
#include <hip/hip_runtime.h>
#include <hip/hip_bf16.h>
#include <math.h>

// S5-SSM forward for MI455X (gfx1250, wave32).
// Dominant cost: conv1d == GEMM 4096x1024x4096 (~34 GFLOP) -> v_wmma_f32_16x16x32_f16.
// Conv A-operand staged in LDS via async global->LDS copies (ASYNCcnt path);
// everything else is bandwidth-bound (~150MB intermediates, L2-resident at 192MB).

typedef __attribute__((ext_vector_type(16))) _Float16 v16h;
typedef __attribute__((ext_vector_type(8)))  _Float16 v8h;
typedef __attribute__((ext_vector_type(8)))  float    v8f;

#define LSEQ   4096
#define EPHC   1024
#define CHUNK  128
#define NCHUNK 32
#define SROWS  19            // 16 M rows + (K-1)=3 halo rows
#define SSTRIDE 1032         // 1024 + 8 halves pad (2064B/row, 16B aligned)

// ---- WMMA fragment helper (ISA 7.12.2 16-bit A-matrix 16x32 layout, wave32) ----
// lane m = lane&15; kb = (lane>>4)*8; elems 0..7 -> K=kb..kb+7; 8..15 -> K=kb+16..kb+23.
__device__ __forceinline__ v16h load_frag_a(const _Float16* __restrict__ base,
                                            int row0, int lda, int k0, int lane) {
  int m  = lane & 15;
  int kb = (lane >> 4) << 3;
  const _Float16* pr = base + (size_t)(row0 + m) * lda + k0 + kb;
  v8h lo = *(const v8h*)pr;
  v8h hi = *(const v8h*)(pr + 16);
  v16h r;
#pragma unroll
  for (int i = 0; i < 8; ++i) { r[i] = lo[i]; r[i + 8] = hi[i]; }
  return r;
}

__device__ __forceinline__ float siluf(float v) { return v / (1.f + __expf(-v)); }

// ---- prep: per-channel ZOH constants A0 = exp(Lambda*step), Bc = (A0-1)/Lambda ----
__global__ void prep_const_kernel(const float* __restrict__ Lre, const float* __restrict__ Lim,
                                  const float* __restrict__ log_step,
                                  float2* __restrict__ A0, float2* __restrict__ Bc) {
  int i = blockIdx.x * blockDim.x + threadIdx.x;
  if (i >= EPHC) return;
  int h = i >> 5;
  float st  = __expf(log_step[h]);
  float zr  = Lre[i] * st, zi = Lim[i] * st;
  float er  = __expf(zr);
  float a0r = er * __cosf(zi), a0i = er * __sinf(zi);
  A0[i] = make_float2(a0r, a0i);
  float lr = Lre[i], li = Lim[i];
  float inv = 1.f / (lr * lr + li * li);
  float nr = a0r - 1.f, ni = a0i;
  Bc[i] = make_float2((nr * lr + ni * li) * inv, (ni * lr - nr * li) * inv);
}

__global__ void cvt_x_kernel(const float* __restrict__ x, _Float16* __restrict__ xh, int n) {
  int i = blockIdx.x * blockDim.x + threadIdx.x;
  if (i < n) xh[i] = (_Float16)x[i];
}

// zero the conv 'SAME' pad rows of B_proj_f16: row 0 (l=-1) and rows L+1, L+2 (l=L, L+1)
__global__ void zero_pad_kernel(_Float16* __restrict__ Bp) {
  int i = blockIdx.x * blockDim.x + threadIdx.x;
  if (i < EPHC) Bp[i] = (_Float16)0.f;
  else if (i < 3 * EPHC) Bp[(size_t)(LSEQ + 1) * EPHC + (i - EPHC)] = (_Float16)0.f;
}

// ---- swizzle fp32 KxN weights into WMMA B-fragment order (one 32B load / lane) ----
__global__ void swizzleB_kernel(const float* __restrict__ W, _Float16* __restrict__ dst, int N) {
  int tile = blockIdx.x;              // kt * (N/16) + nt
  int lane = threadIdx.x;
  int ntc  = N >> 4;
  int kt = tile / ntc, nt = tile - kt * ntc;
  int n  = (nt << 4) + (lane & 15);
  int kb = (kt << 5) + ((lane >> 4) << 3);
  v16h v;
#pragma unroll
  for (int e = 0; e < 16; ++e) {
    int k = kb + e + ((e >= 8) ? 8 : 0);
    v[e] = (_Float16)W[(size_t)k * N + n];
  }
  ((v16h*)dst)[(size_t)tile * 32 + lane] = v;
}

// ---- generic WMMA GEMM: out = A(Mx32K) @ Bsw + bias, optional silu, f32/f16 stores ----
__global__ void gemm_wmma_kernel(const _Float16* __restrict__ A16, int lda,
                                 const _Float16* __restrict__ Bsw, int NT, int Ksteps,
                                 const float* __restrict__ bias,
                                 float* __restrict__ outF, _Float16* __restrict__ outH,
                                 int ldo, int do_silu) {
  int lane = threadIdx.x & 31;
  int tile = blockIdx.x * 4 + (threadIdx.x >> 5);
  int mt = tile / NT, nt = tile - mt * NT;
  v8f acc = {};
  for (int kt = 0; kt < Ksteps; ++kt) {
    v16h a = load_frag_a(A16, mt << 4, lda, kt << 5, lane);
    v16h b = ((const v16h*)Bsw)[((size_t)(kt * NT + nt) << 5) + lane];
    acc = __builtin_amdgcn_wmma_f32_16x16x32_f16(false, a, false, b, (short)0, acc, false, false);
  }
  int n    = (nt << 4) + (lane & 15);
  int mrow = (mt << 4) + ((lane >> 4) << 3);
  float bv = bias ? bias[n] : 0.f;
#pragma unroll
  for (int r = 0; r < 8; ++r) {
    float v = acc[r] + bv;
    if (do_silu) v = siluf(v);
    size_t o = (size_t)(mrow + r) * ldo + n;
    if (outF) outF[o] = v;
    if (outH) outH[o] = (_Float16)v;
  }
}

// ---- conv1d (K=4, SAME pad (1,2)) as WMMA GEMM, A staged in LDS via async copies ----
// epilogue: B_act = silu(conv+b); A = log(1+exp(A0 + LamAct)) (complex);
//           Bu = Bc * B_tilde * B_act (complex)
__global__ void conv_scanprep_kernel(const _Float16* __restrict__ Bp,   // (L+3) x 1024, row0 = l=-1
                                     const _Float16* __restrict__ Wsw,  // 128 ktiles x 64 ntiles
                                     const float* __restrict__ conv_b,
                                     const float* __restrict__ Lam_act,
                                     const float* __restrict__ Bmat,    // (L, 32, 2)
                                     const float2* __restrict__ A0,
                                     const float2* __restrict__ Bc,
                                     float2* __restrict__ A,
                                     float2* __restrict__ Bu) {
  __shared__ _Float16 sA[SROWS * SSTRIDE];   // ~38.7 KB
  const int NT = EPHC >> 4;  // 64
  int tid  = threadIdx.x;
  int lane = tid & 31;
  int wave = tid >> 5;
  int mt   = blockIdx.x >> 4;          // 0..255
  int nt   = ((blockIdx.x & 15) << 2) + wave;

  // Stage 19 rows (padded rows mt*16 .. mt*16+18) of B_proj_f16 into LDS.
  // 128 threads x 16B cover one 2048B row; async global->LDS (ASYNCcnt).
  {
    const _Float16* g = Bp + ((size_t)(mt << 4)) * EPHC + (tid << 3);
    unsigned lbase = (unsigned)(size_t)(void*)sA + (unsigned)(tid << 4);
#pragma unroll
    for (int r = 0; r < SROWS; ++r) {
      unsigned    loff = lbase + (unsigned)(r * (SSTRIDE * 2));
      const void* gp   = (const void*)(g + (size_t)r * EPHC);
      asm volatile("global_load_async_to_lds_b128 %0, %1, off"
                   :: "v"(loff), "v"(gp) : "memory");
    }
    asm volatile("s_wait_asynccnt 0x0" ::: "memory");
    __syncthreads();
  }

  v8f acc = {};
  for (int t = 0; t < 4; ++t) {        // conv tap: padded row (l + t - 1) + 1 -> local row t + m
#pragma unroll 4
    for (int kc = 0; kc < 32; ++kc) {
      int kt = (t << 5) + kc;
      const v16h* bptr = ((const v16h*)Wsw) + ((size_t)(kt * NT + nt) << 5) + lane;
      __builtin_prefetch((const void*)(bptr + (NT << 5)), 0, 1);  // next k-tile weights
      v16h a = load_frag_a(sA, t, SSTRIDE, kc << 5, lane);        // ds_load_b128 x2
      v16h b = *bptr;
      acc = __builtin_amdgcn_wmma_f32_16x16x32_f16(false, a, false, b, (short)0, acc, false, false);
    }
  }

  int n    = (nt << 4) + (lane & 15);     // channel ch = h*32 + p
  int mrow = (mt << 4) + ((lane >> 4) << 3);
  int pcol = n & 31;
  float  cbv = conv_b[n];
  float2 a0  = A0[n];
  float2 bc  = Bc[n];
#pragma unroll
  for (int r = 0; r < 8; ++r) {
    int l = mrow + r;
    size_t i = ((size_t)l << 10) + n;
    float ba  = siluf(acc[r] + cbv);
    float lam = Lam_act[i];
    float zr = a0.x + lam, zi = a0.y;
    float er = __expf(zr);
    float wr = er * __cosf(zi), wi = er * __sinf(zi);
    float u  = 1.f + wr;
    float2 Av = make_float2(0.5f * __logf(u * u + wi * wi), atan2f(wi, u));
    float btr = Bmat[((size_t)l << 6) + (pcol << 1)];
    float bti = Bmat[((size_t)l << 6) + (pcol << 1) + 1];
    float2 bu = make_float2((bc.x * btr - bc.y * bti) * ba,
                            (bc.x * bti + bc.y * btr) * ba);
    A[i]  = Av;
    Bu[i] = bu;
  }
}

// ---- 3-pass chunked complex scan: x_l = A_l * x_{l-1} + Bu_l, in-place ----
__global__ void scan_pass1_kernel(float2* __restrict__ A, float2* __restrict__ Bu) {
  int t = blockIdx.x * blockDim.x + threadIdx.x;   // 1024 ch x 32 chunks
  int ch = t & (EPHC - 1), c = t >> 10;
  size_t base = ((size_t)c * CHUNK) << 10;
  float2 ap = make_float2(1.f, 0.f);
  float2 xx = make_float2(0.f, 0.f);
  for (int j = 0; j < CHUNK; ++j) {
    size_t idx = base + ((size_t)j << 10) + ch;
    float2 a = A[idx], b = Bu[idx];
    float2 nx = make_float2(a.x * xx.x - a.y * xx.y + b.x, a.x * xx.y + a.y * xx.x + b.y);
    float2 np = make_float2(a.x * ap.x - a.y * ap.y,       a.x * ap.y + a.y * ap.x);
    xx = nx; ap = np;
    A[idx] = ap;   // A-prefix within chunk
    Bu[idx] = xx;  // local scan value
  }
}

__global__ void scan_pass2_kernel(const float2* __restrict__ A, const float2* __restrict__ Bu,
                                  float2* __restrict__ carries) {
  int ch = blockIdx.x * blockDim.x + threadIdx.x;
  if (ch >= EPHC) return;
  float2 cr = make_float2(0.f, 0.f);
  for (int c = 0; c < NCHUNK; ++c) {
    carries[(size_t)c * EPHC + ch] = cr;
    size_t last = (((size_t)c * CHUNK + CHUNK - 1) << 10) + ch;
    float2 ap = A[last], xl = Bu[last];
    cr = make_float2(ap.x * cr.x - ap.y * cr.y + xl.x, ap.x * cr.y + ap.y * cr.x + xl.y);
  }
}

__global__ void scan_pass3_kernel(const float2* __restrict__ A, float2* __restrict__ Bu,
                                  const float2* __restrict__ carries) {
  size_t i = (size_t)blockIdx.x * blockDim.x + threadIdx.x;
  int ch = (int)(i & (EPHC - 1));
  int l  = (int)(i >> 10);
  int c  = l / CHUNK;
  float2 cr = carries[(size_t)c * EPHC + ch];
  float2 ap = A[i], xl = Bu[i];
  Bu[i] = make_float2(xl.x + ap.x * cr.x - ap.y * cr.y,
                      xl.y + ap.x * cr.y + ap.y * cr.x);
}

// ---- ys (handles the reference's raw reshape: ys[l,h] = Re sum_j C[l,j]*xs[l,j,h]) + gate ----
__global__ void ys_gate_kernel(const float2* __restrict__ xs, const float* __restrict__ Cmat,
                               const float* __restrict__ x, const float* __restrict__ gt,
                               const float* __restrict__ Wd, const float* __restrict__ bd,
                               _Float16* __restrict__ ht16) {
  int tid = threadIdx.x;
  int l = (blockIdx.x << 2) + (tid >> 5);
  int h = tid & 31;
  const float2* xr = xs + ((size_t)l << 10);
  const float*  cl = Cmat + ((size_t)l << 6);
  float ys = 0.f;
#pragma unroll
  for (int j = 0; j < 32; ++j) {
    float2 xv = xr[(j << 5) + h];
    ys += cl[j << 1] * xv.x - cl[(j << 1) + 1] * xv.y;
  }
  const float* xl = x + ((size_t)l << 5);
  float accd = bd[0];
#pragma unroll
  for (int k = 0; k < 32; ++k) accd += xl[k] * Wd[k];
  float dp = __logf(1.f + __expf(accd));       // softplus
  float g  = gt[((size_t)l << 5) + h];
  float xv = xl[h];
  float htv = (1.f - g) * (ys + dp * xv) + g * xv;
  ht16[((size_t)l << 5) + h] = (_Float16)htv;
}

extern "C" void kernel_launch(void* const* d_in, const int* in_sizes, int n_in,
                              void* d_out, int out_size, void* d_ws, size_t ws_size,
                              hipStream_t stream) {
  (void)in_sizes; (void)n_in; (void)out_size; (void)ws_size;
  const float* x   = (const float*)d_in[0];
  const float* Lre = (const float*)d_in[1];
  const float* Lim = (const float*)d_in[2];
  const float* Bm  = (const float*)d_in[3];
  const float* Cm  = (const float*)d_in[4];
  const float* lst = (const float*)d_in[5];
  const float* Wb  = (const float*)d_in[6];
  const float* bb  = (const float*)d_in[7];
  const float* Wl  = (const float*)d_in[8];
  const float* bl  = (const float*)d_in[9];
  const float* cw  = (const float*)d_in[10];
  const float* cb  = (const float*)d_in[11];
  const float* Wg  = (const float*)d_in[12];
  const float* bg  = (const float*)d_in[13];
  const float* Wd  = (const float*)d_in[14];
  const float* bd  = (const float*)d_in[15];
  const float* Wo  = (const float*)d_in[16];
  const float* bo  = (const float*)d_in[17];
  float* outp = (float*)d_out;

  char* p = (char*)d_ws;
  auto alloc = [&](size_t bytes) -> char* {
    char* r = p; p += (bytes + 255) & ~((size_t)255); return r;
  };
  _Float16* x16    = (_Float16*)alloc((size_t)LSEQ * 32 * 2);
  _Float16* Wb_sw  = (_Float16*)alloc(64u * 1024);
  _Float16* Wl_sw  = (_Float16*)alloc(64u * 1024);
  _Float16* Wg_sw  = (_Float16*)alloc(2u * 1024);
  _Float16* Wo_sw  = (_Float16*)alloc(2u * 1024);
  _Float16* CW_sw  = (_Float16*)alloc(8192u * 1024);                  // 8.4 MB
  _Float16* Bp     = (_Float16*)alloc((size_t)(LSEQ + 3) * EPHC * 2); // 8.4 MB padded
  float*    Lam    = (float*)   alloc((size_t)LSEQ * EPHC * 4);       // 16.8 MB
  float2*   A      = (float2*)  alloc((size_t)LSEQ * EPHC * 8);       // 33.6 MB
  float2*   Bu     = (float2*)  alloc((size_t)LSEQ * EPHC * 8);       // 33.6 MB
  float*    gt     = (float*)   alloc((size_t)LSEQ * 32 * 4);
  _Float16* ht16   = (_Float16*)alloc((size_t)LSEQ * 32 * 2);
  float2*   A0     = (float2*)  alloc(EPHC * 8);
  float2*   Bc     = (float2*)  alloc(EPHC * 8);
  float2*   carr   = (float2*)  alloc((size_t)NCHUNK * EPHC * 8);

  // prep
  prep_const_kernel<<<4, 256, 0, stream>>>(Lre, Lim, lst, A0, Bc);
  cvt_x_kernel<<<512, 256, 0, stream>>>(x, x16, LSEQ * 32);
  zero_pad_kernel<<<12, 256, 0, stream>>>(Bp);
  swizzleB_kernel<<<64, 32, 0, stream>>>(Wb, Wb_sw, EPHC);
  swizzleB_kernel<<<64, 32, 0, stream>>>(Wl, Wl_sw, EPHC);
  swizzleB_kernel<<<2, 32, 0, stream>>>(Wg, Wg_sw, 32);
  swizzleB_kernel<<<2, 32, 0, stream>>>(Wo, Wo_sw, 32);
  swizzleB_kernel<<<8192, 32, 0, stream>>>(cw, CW_sw, EPHC);  // conv_w viewed (4096,1024)

  // input projections: x @ Wb -> f16 (padded rows 1..L), x @ Wl -> silu f32, x @ Wg -> silu f32
  gemm_wmma_kernel<<<4096, 128, 0, stream>>>(x16, 32, Wb_sw, 64, 1, bb, nullptr, Bp + EPHC, EPHC, 0);
  gemm_wmma_kernel<<<4096, 128, 0, stream>>>(x16, 32, Wl_sw, 64, 1, bl, Lam, nullptr, EPHC, 1);
  gemm_wmma_kernel<<<128, 128, 0, stream>>>(x16, 32, Wg_sw, 2, 1, bg, gt, nullptr, 32, 1);

  // conv GEMM (LDS-staged A) + fused scan-input math
  conv_scanprep_kernel<<<4096, 128, 0, stream>>>(Bp, CW_sw, cb, Lam, Bm, A0, Bc, A, Bu);

  // complex associative scan
  scan_pass1_kernel<<<128, 256, 0, stream>>>(A, Bu);
  scan_pass2_kernel<<<4, 256, 0, stream>>>(A, Bu, carr);
  scan_pass3_kernel<<<16384, 256, 0, stream>>>(A, Bu, carr);

  // output mix + gate, then ht @ Wo + bo -> out (fp32)
  ys_gate_kernel<<<1024, 128, 0, stream>>>(Bu, Cm, x, gt, Wd, bd, ht16);
  gemm_wmma_kernel<<<128, 128, 0, stream>>>(ht16, 32, Wo_sw, 2, 1, bo, outp, nullptr, 32, 0);
}
